// LSTMOptimizer_19447611916789
// MI455X (gfx1250) — compile-verified, matrix operand order checked by reading, not run on hardware
//
#include <hip/hip_runtime.h>
#include <hip/hip_bf16.h>

typedef _Float16 half_t;
typedef __attribute__((ext_vector_type(16))) _Float16 v16h;
typedef __attribute__((ext_vector_type(4)))  _Float16 v4h;
typedef __attribute__((ext_vector_type(8)))  float    v8f;
typedef __attribute__((ext_vector_type(2)))  float    v2f;

#define NPAR 524288
#define H    128
#define GH   512   // 4*H
#define MT   2     // M-tiles per wave (32 rows)
#define ROWS_PER_BLOCK 256

union AFrag { v16h v; _Float16 e[16]; };
union Acc   { v8f  v; float    e[8];  };

__global__ __launch_bounds__(256) void lstm_opt_wmma(
    const float* __restrict__ param, const float* __restrict__ grad,
    const float* __restrict__ hin,   const float* __restrict__ cin,
    const float* __restrict__ momentum, const float* __restrict__ prev_update,
    const float* __restrict__ W_ih,  const float* __restrict__ W_hh,
    const float* __restrict__ b_ih,  const float* __restrict__ b_hh,
    const float* __restrict__ W_up,  const float* __restrict__ b_up,
    float* __restrict__ out_update, float* __restrict__ out_h,
    float* __restrict__ out_c,      float* __restrict__ out_mom)
{
    extern __shared__ char smem_raw[];
    half_t* Wlds  = (half_t*)smem_raw;                               // [512][128] f16, 128KB
    half_t* hTile = (half_t*)(smem_raw + (size_t)GH * H * 2);        // [8*32][128] f16, 64KB
    float*  WihL  = (float*)(smem_raw + (size_t)GH * H * 2 + (size_t)ROWS_PER_BLOCK * H * 2);
    float*  bsum  = WihL + GH * 4;                                   // [512]
    float*  WupL  = bsum + GH;                                       // [128]

    const int tid   = threadIdx.x;
    const int lane  = tid & 31;
    const int wave  = tid >> 5;
    const int row0  = blockIdx.x * ROWS_PER_BLOCK + wave * 32;  // first of 32 rows this wave owns

    // ---- cooperative staging (global_load_b128 + ds_store_b64) ----
    {
        const float4* W4 = (const float4*)W_hh;
        for (int i = tid; i < GH * H / 4; i += 256) {
            float4 w = W4[i];
            v4h hv;
            hv.x = (half_t)w.x; hv.y = (half_t)w.y;
            hv.z = (half_t)w.z; hv.w = (half_t)w.w;
            *(v4h*)&Wlds[i * 4] = hv;
        }
    }
    for (int i = tid; i < GH * 4; i += 256) WihL[i] = W_ih[i];
    for (int i = tid; i < GH;     i += 256) bsum[i] = b_ih[i] + b_hh[i];
    if (tid < H) WupL[tid] = W_up[tid];
    // wave's 32x128 h tile -> f16 LDS (rows contiguous => linear copy of 4096 floats)
    {
        const float4* h4 = (const float4*)(hin + (size_t)row0 * H);
        for (int i = lane; i < 32 * H / 4; i += 32) {
            float4 w = h4[i];
            v4h hv;
            hv.x = (half_t)w.x; hv.y = (half_t)w.y;
            hv.z = (half_t)w.z; hv.w = (half_t)w.w;
            *(v4h*)&hTile[wave * 32 * H + i * 4] = hv;
        }
    }
    __syncthreads();

    // ---- A fragments (f16 16x32) for both M-tiles, reused across all gate tiles ----
    // lane L: M = L&15 ; K halves: [ko..ko+7], [ko+16..ko+23], ko = (L>=16)?8:0
    const int mA = lane & 15;
    const int ko = (lane >> 4) << 3;
    AFrag afr[MT][4];
#pragma unroll
    for (int mt = 0; mt < MT; ++mt) {
#pragma unroll
        for (int kk = 0; kk < 4; ++kk) {
            const half_t* src = &hTile[wave * 32 * H + (mt * 16 + mA) * H + kk * 32 + ko];
#pragma unroll
            for (int t = 0; t < 8; ++t) {
                afr[mt][kk].e[t]     = src[t];
                afr[mt][kk].e[8 + t] = src[16 + t];
            }
        }
    }

    // C/D layout: VGPR r <-> (M = r + mOff, N = lane&15)
    const int nlane = lane & 15;
    const int mOff  = (lane >> 4) << 3;

#if __has_builtin(__builtin_amdgcn_wmma_f32_16x16x4_f32)
    // f32 WMMA A-fragments for x @ W_ih^T : A is 16x4 (M=lane&15; K pair by lane half)
    // x feature order: (grad, param, momentum, prev_update); pointer-select keeps EXEC full
    const float* xs0 = (lane < 16) ? grad  : momentum;
    const float* xs1 = (lane < 16) ? param : prev_update;
    v2f aInit[MT];
#pragma unroll
    for (int mt = 0; mt < MT; ++mt) {
        aInit[mt].x = xs0[row0 + mt * 16 + nlane];
        aInit[mt].y = xs1[row0 + mt * 16 + nlane];
    }
    const int kbase = (lane >> 4) << 1;   // 0 or 2
#else
    float xg[MT][8], xp[MT][8], xm[MT][8], xu[MT][8];
#pragma unroll
    for (int mt = 0; mt < MT; ++mt)
#pragma unroll
        for (int r = 0; r < 8; ++r) {
            const int p = row0 + mt * 16 + mOff + r;
            xg[mt][r] = grad[p]; xp[mt][r] = param[p];
            xm[mt][r] = momentum[p]; xu[mt][r] = prev_update[p];
        }
#endif

    float pu[MT][8];
#pragma unroll
    for (int mt = 0; mt < MT; ++mt)
#pragma unroll
        for (int r = 0; r < 8; ++r) pu[mt][r] = 0.0f;

    // ---- main loop: hidden-column blocks of 16; gates (i,f,g,o) together; 2 M-tiles ----
    for (int j = 0; j < 8; ++j) {
        Acc acc[MT][4];

        // C init = (b_ih + b_hh) broadcast, then x @ W_ih^T via f32 WMMA
        // (B-fragment of W_ih shared by both M-tiles)
#pragma unroll
        for (int gsel = 0; gsel < 4; ++gsel) {
            const int gcol = gsel * H + j * 16 + nlane;
            const float bb = bsum[gcol];
#pragma unroll
            for (int mt = 0; mt < MT; ++mt)
#pragma unroll
                for (int r = 0; r < 8; ++r) acc[mt][gsel].e[r] = bb;
#if __has_builtin(__builtin_amdgcn_wmma_f32_16x16x4_f32)
            v2f bInit;   // B 4x16: lane = N col; K = kbase, kbase+1
            bInit.x = WihL[gcol * 4 + kbase];
            bInit.y = WihL[gcol * 4 + kbase + 1];
#pragma unroll
            for (int mt = 0; mt < MT; ++mt)
                acc[mt][gsel].v = __builtin_amdgcn_wmma_f32_16x16x4_f32(
                    false, aInit[mt], false, bInit, (short)0, acc[mt][gsel].v, false, false);
#else
            const float w0 = WihL[gcol * 4 + 0], w1 = WihL[gcol * 4 + 1];
            const float w2 = WihL[gcol * 4 + 2], w3 = WihL[gcol * 4 + 3];
#pragma unroll
            for (int mt = 0; mt < MT; ++mt)
#pragma unroll
                for (int r = 0; r < 8; ++r)
                    acc[mt][gsel].e[r] += xg[mt][r] * w0 + xp[mt][r] * w1
                                        + xm[mt][r] * w2 + xu[mt][r] * w3;
#endif
        }

        // f16 WMMA over K = 128 (4 steps); each B-fragment feeds both M-tiles
#pragma unroll
        for (int gsel = 0; gsel < 4; ++gsel) {
            const int n0 = gsel * H + j * 16;
#pragma unroll
            for (int kk = 0; kk < 4; ++kk) {
                AFrag bfr;
                const half_t* bsrc = &Wlds[(n0 + nlane) * H + kk * 32 + ko];
#pragma unroll
                for (int t = 0; t < 8; ++t) {
                    bfr.e[t]     = bsrc[t];
                    bfr.e[8 + t] = bsrc[16 + t];
                }
#pragma unroll
                for (int mt = 0; mt < MT; ++mt)
                    acc[mt][gsel].v = __builtin_amdgcn_wmma_f32_16x16x32_f16(
                        false, afr[mt][kk].v, false, bfr.v, (short)0,
                        acc[mt][gsel].v, false, false);
            }
        }

        // elementwise LSTM cell + head partials
        const int   hcol = j * 16 + nlane;
        const float wup  = WupL[hcol];
#pragma unroll
        for (int mt = 0; mt < MT; ++mt) {
#pragma unroll
            for (int r = 0; r < 8; ++r) {
                const int p = row0 + mt * 16 + mOff + r;
                const float ig = 1.0f / (1.0f + __expf(-acc[mt][0].e[r]));
                const float fg = 1.0f / (1.0f + __expf(-acc[mt][1].e[r]));
                const float gg = tanhf(acc[mt][2].e[r]);
                const float og = 1.0f / (1.0f + __expf(-acc[mt][3].e[r]));
                const size_t idx = (size_t)p * H + hcol;
                const float cold = cin[idx];
                const float cn = fg * cold + ig * gg;
                const float hn = og * tanhf(cn);
                out_c[idx] = cn;
                out_h[idx] = hn;
                pu[mt][r] += hn * wup;
            }
        }
    }

    // ---- reduce update over the 16 hidden-lanes in each half-wave, write scalars ----
    const float bup = b_up[0];
#pragma unroll
    for (int mt = 0; mt < MT; ++mt) {
#pragma unroll
        for (int r = 0; r < 8; ++r) {
            float v = pu[mt][r];
            v += __shfl_xor(v, 1);
            v += __shfl_xor(v, 2);
            v += __shfl_xor(v, 4);
            v += __shfl_xor(v, 8);
            if (nlane == 0) {
                const int p = row0 + mt * 16 + mOff + r;
                const float upd = v + bup;
                out_update[p] = upd;
                out_mom[p]    = 0.9f * momentum[p] + upd;
            }
        }
    }
}

extern "C" void kernel_launch(void* const* d_in, const int* in_sizes, int n_in,
                              void* d_out, int out_size, void* d_ws, size_t ws_size,
                              hipStream_t stream) {
    const float* param       = (const float*)d_in[0];
    const float* grad        = (const float*)d_in[1];
    const float* h           = (const float*)d_in[2];
    const float* c           = (const float*)d_in[3];
    const float* momentum    = (const float*)d_in[4];
    const float* prev_update = (const float*)d_in[5];
    const float* W_ih        = (const float*)d_in[6];
    const float* W_hh        = (const float*)d_in[7];
    const float* b_ih        = (const float*)d_in[8];
    const float* b_hh        = (const float*)d_in[9];
    const float* W_up        = (const float*)d_in[10];
    const float* b_up        = (const float*)d_in[11];

    float* out        = (float*)d_out;
    float* out_update = out;
    float* out_h      = out + (size_t)NPAR;
    float* out_c      = out + (size_t)NPAR + (size_t)NPAR * H;
    float* out_mom    = out + (size_t)NPAR + 2 * (size_t)NPAR * H;

    const size_t lds_bytes = (size_t)GH * H * 2                 // W_hh f16     (128KB)
                           + (size_t)ROWS_PER_BLOCK * H * 2     // h tiles f16  (64KB)
                           + (size_t)GH * 4 * 4                 // W_ih f32
                           + (size_t)GH * 4                     // bias sum
                           + (size_t)H * 4;                     // W_up

    dim3 grid(NPAR / ROWS_PER_BLOCK), block(256);
    hipLaunchKernelGGL(lstm_opt_wmma, grid, block, lds_bytes, stream,
                       param, grad, h, c, momentum, prev_update,
                       W_ih, W_hh, b_ih, b_hh, W_up, b_up,
                       out_update, out_h, out_c, out_mom);
}
